// WA_3GConv_22943715295614
// MI455X (gfx1250) — matrix operand order repaired and checked by reading, not compiled, not run on hardware
//
#include <hip/hip_runtime.h>
#include <hip/hip_bf16.h>
#include <math.h>

// ---------------- problem constants ----------------
constexpr int B_    = 16;
constexpr int INP   = 256;
constexpr int OUP   = 256;
constexpr int H_    = 64;
constexpr int W_    = 64;
constexpr int CGRP  = 8;
constexpr int ICG   = INP / CGRP;     // 32 input channels per group
constexpr int OCG   = OUP / CGRP;     // 32 output channels per group
constexpr int KK    = ICG * 9;        // 288 total reduction length
constexpr int RPB   = 4;              // output rows per conv workgroup
constexpr int ICP   = 40;             // padded channel dim in LDS (80B stride)

typedef __attribute__((ext_vector_type(16))) _Float16 v16h;
typedef __attribute__((ext_vector_type(8)))  _Float16 v8h;   // 16-byte fragment piece
typedef __attribute__((ext_vector_type(8)))  float    v8f;

// =====================================================================
// Kernel 0: one-pass f32->f16 swizzle to channel-innermost layout
//   xsw[bg][y][x][ic]  (bg = b*8+g, ic in [0,32))
// fused with per-(bg,y,ic) row sums for the GAP reduction (deterministic).
// One block per (bg, y).
// =====================================================================
__global__ __launch_bounds__(256)
void swizzle_gap_kernel(const float* __restrict__ x,
                        _Float16* __restrict__ xsw,
                        float* __restrict__ psum) {
  const int blk = blockIdx.x;          // 0 .. 8191
  const int y   = blk & 63;
  const int bg  = blk >> 6;            // b*8+g
  const int tid = threadIdx.x;

  __shared__ float tile[ICG][W_ + 1];  // +1 pad: avoid bank conflicts

  // coalesced read of 32 channel-rows of width 64
  for (int idx = tid; idx < ICG * W_; idx += 256) {
    const int ic   = idx >> 6;
    const int xcol = idx & 63;
    tile[ic][xcol] = x[((size_t)(bg * ICG + ic) * H_ + y) * W_ + xcol];
  }
  __syncthreads();

  // per-channel row sums (32 threads)
  if (tid < ICG) {
    float s = 0.f;
    #pragma unroll 8
    for (int xc = 0; xc < W_; ++xc) s += tile[tid][xc];
    psum[(size_t)(bg * H_ + y) * ICG + tid] = s;
  }

  // pack: one 16B store per thread, fully coalesced (4 KB per row)
  {
    const int xcol = tid >> 2;
    const int q    = tid & 3;
    v8h pk;
    #pragma unroll
    for (int j = 0; j < 8; ++j) pk[j] = (_Float16)tile[q * 8 + j][xcol];
    _Float16* dst = xsw + ((size_t)bg * H_ + y) * ((size_t)W_ * ICG)
                        + (size_t)xcol * ICG + q * 8;
    *(v8h*)dst = pk;
  }
}

// =====================================================================
// Kernel 1: finish GAP: reduce psum over y (deterministic), scale by 1/HW
// =====================================================================
__global__ __launch_bounds__(256)
void gap_reduce_kernel(const float* __restrict__ psum, float* __restrict__ gap) {
  const int b = blockIdx.x;            // 0..15
  const int c = threadIdx.x;           // 0..255
  const int g  = c >> 5;
  const int ic = c & 31;
  const int bg = b * CGRP + g;
  float s = 0.f;
  for (int y = 0; y < H_; ++y)
    s += psum[(size_t)(bg * H_ + y) * ICG + ic];
  gap[b * INP + c] = s * (1.0f / (H_ * W_));
}

// =====================================================================
// Kernel 2: both GLGF heads from gap.
// =====================================================================
__global__ __launch_bounds__(256)
void glgf_kernel(const float* __restrict__ gap,
                 const float* __restrict__ g1w1l,  const float* __restrict__ g1w1g1,
                 const float* __restrict__ g1w1g2, const float* __restrict__ g1w2g,
                 const float* __restrict__ g2w1l,  const float* __restrict__ g2w1g1,
                 const float* __restrict__ g2w1g2, const float* __restrict__ g2w2l,
                 const float* __restrict__ g2w2g,
                 float* __restrict__ h1out, float* __restrict__ g2b,
                 float* __restrict__ cg) {
  const int b = blockIdx.x;
  const int i = threadIdx.x;
  const int gi = i >> 4;
  __shared__ float g[256];
  __shared__ float tg[16];
  __shared__ float red[256];

  g[i] = gap[b * INP + i];
  __syncthreads();

  // ---- GLGF head 1 (dynamic-weight generator) ----
  float local1 = 0.f;
  #pragma unroll
  for (int ii = 0; ii < 16; ++ii) local1 += g[gi * 16 + ii] * g1w1l[i * 16 + ii];
  if (i < 16) {
    float t = 0.f;
    #pragma unroll
    for (int ii = 0; ii < 16; ++ii) t += g[i * 16 + ii] * g1w1g1[i * 16 + ii];
    tg[i] = t;
  }
  __syncthreads();
  float glob1 = 0.f;
  #pragma unroll
  for (int gg = 0; gg < 16; ++gg) glob1 += tg[gg] * g1w1g2[i * 16 + gg];
  const float h1v = 1.f / (1.f + expf(-(local1 + glob1)));

  red[i] = h1v * g1w2g[i];
  __syncthreads();
  for (int off = 128; off > 0; off >>= 1) {
    if (i < off) red[i] += red[i + off];
    __syncthreads();
  }
  if (i == 0) g2b[b] = red[0];
  h1out[b * INP + i] = h1v;
  __syncthreads();

  // ---- GLGF head 2 (output gate) ----
  float local2 = 0.f;
  #pragma unroll
  for (int ii = 0; ii < 16; ++ii) local2 += g[gi * 16 + ii] * g2w1l[i * 16 + ii];
  if (i < 16) {
    float t = 0.f;
    #pragma unroll
    for (int ii = 0; ii < 16; ++ii) t += g[i * 16 + ii] * g2w1g1[i * 16 + ii];
    tg[i] = t;
  }
  __syncthreads();
  float glob2 = 0.f;
  #pragma unroll
  for (int gg = 0; gg < 16; ++gg) glob2 += tg[gg] * g2w1g2[i * 16 + gg];
  const float h2v = 1.f / (1.f + expf(-(local2 + glob2)));

  red[i] = h2v * g2w2g[i];
  __syncthreads();
  for (int off = 128; off > 0; off >>= 1) {
    if (i < off) red[i] += red[i + off];
    __syncthreads();
  }
  cg[b * OUP + i] = h2v * g2w2l[i] + red[0];
}

// =====================================================================
// Kernel 3: dynamic grouped 3x3 conv as 9 shifted WMMA GEMMs (K=32 each).
//  x tile staged via async global->LDS b128 copies from the pre-swizzled
//  f16 tensor (overlapped with dynamic-weight expansion), then 36
//  v_wmma_f32_16x16x32_f16 per wave with contiguous 16B LDS fragments.
// =====================================================================
__global__ __launch_bounds__(256)
void dynconv_wmma_kernel(const _Float16* __restrict__ xsw,
                         const float* __restrict__ w2l,     // g1_fc2_l [73728]
                         const float* __restrict__ h1,      // [B_,256]
                         const float* __restrict__ g2b,     // [B_]
                         const float* __restrict__ cgate,   // [B_,256]
                         float* __restrict__ out) {
  const int blk = blockIdx.x;
  const int rb  = blk & 15;          // row block 0..15
  const int g   = (blk >> 4) & 7;    // channel group 0..7
  const int b   = blk >> 7;          // sample 0..15
  const int y0  = rb * RPB;

  const int tid  = threadIdx.x;
  const int wave = tid >> 5;
  const int lane = tid & 31;

  __shared__ _Float16 wlds[9][OCG][ICP];            // 23.0 KB
  __shared__ _Float16 xlds[RPB + 2][W_ + 2][ICP];   // 31.7 KB

  __builtin_prefetch(&w2l[(size_t)(g * OCG) * KK], 0, 3);

  // ---- Phase B (issue first, completes asynchronously): x tile --------
  // unit = (r, cc, quarter): 6*66*4 = 1584 16-byte chunks
  constexpr int XROWS = RPB + 2;
  constexpr int XCOLS = W_ + 2;
  const unsigned long long xbase =
      (unsigned long long)(size_t)(xsw + (size_t)(b * CGRP + g) * ((size_t)H_ * W_ * ICG));
  for (int u = tid; u < XROWS * XCOLS * 4; u += 256) {
    const int q  = u & 3;
    const int cc = (u >> 2) % XCOLS;
    const int r  = (u >> 2) / XCOLS;
    const int yi = y0 - 1 + r;
    _Float16* ldsp = &xlds[r][cc][q * 8];
    if (yi >= 0 && yi < H_ && cc >= 1 && cc <= W_) {
      const unsigned lds_off = (unsigned)(size_t)ldsp;
      const unsigned voff =
          (unsigned)(((yi * W_ + (cc - 1)) * ICG + q * 8) * sizeof(_Float16));
      asm volatile("global_load_async_to_lds_b128 %0, %1, %2"
                   :: "v"(lds_off), "v"(voff), "s"(xbase) : "memory");
    } else {
      v8h z = {};
      *(v8h*)ldsp = z;            // zero padding (halo cols / OOB rows)
    }
  }

  // ---- Phase A (overlaps with async copy): expand dyn weights ---------
  // w[b,o,ic,tap] = h1[b,o]*w2l[o*288 + ic*9 + tap] + glob2_1[b]
  const float wbias = g2b[b];
  for (int idx = tid; idx < OCG * KK; idx += 256) {   // linear over w2l slice
    const int ol  = idx / KK;
    const int rem = idx - ol * KK;                    // ic*9 + tap
    const int ic  = rem / 9;
    const int tap = rem - ic * 9;
    const int o   = g * OCG + ol;
    const float wv = h1[b * OUP + o] * w2l[(size_t)o * KK + rem] + wbias;
    wlds[tap][ol][ic] = (_Float16)wv;
  }

  asm volatile("s_wait_asynccnt 0x0" ::: "memory");
  __syncthreads();

  // ---- Phase C: WMMA ---------------------------------------------------
  const int mt = wave & 1;           // M tile (out-ch 16*mt..+15)
  const int nt = wave >> 1;          // N tile (cols 16*nt..+15)
  const int half   = (lane >= 16) ? 1 : 0;
  const int lane16 = lane & 15;
  const int m = mt * 16 + lane16;    // A row (out channel in group)
  const int n = nt * 16 + lane16;    // B col (x column)

  // Hoist all 9 A fragments (row-invariant).
  // A 16x32 f16 layout: half 0 -> K 0-7 & 16-23 ; half 1 -> K 8-15 & 24-31.
  v16h afr[9];
  #pragma unroll
  for (int tap = 0; tap < 9; ++tap) {
    const v8h a0 = *(const v8h*)&wlds[tap][m][half * 8];
    const v8h a1 = *(const v8h*)&wlds[tap][m][16 + half * 8];
    afr[tap] = __builtin_shufflevector(a0, a1,
        0, 1, 2, 3, 4, 5, 6, 7, 8, 9, 10, 11, 12, 13, 14, 15);
  }

  // Gates for epilogue (C/D layout: M = r + 8*half)
  float gates[8];
  #pragma unroll
  for (int r = 0; r < 8; ++r)
    gates[r] = cgate[b * OUP + g * OCG + mt * 16 + half * 8 + r];

  for (int row = 0; row < RPB; ++row) {
    v8f acc = {};
    #pragma unroll
    for (int ky = 0; ky < 3; ++ky) {
      #pragma unroll
      for (int kx = 0; kx < 3; ++kx) {
        // B 32x16 f16 layout: half 0 -> K 0-7 & 16-23 ; half 1 -> K 8-15 & 24-31
        const _Float16* pb = &xlds[row + ky][n + kx][half * 8];
        const v8h b0 = *(const v8h*)pb;
        const v8h b1 = *(const v8h*)(pb + 16);
        const v16h bf = __builtin_shufflevector(b0, b1,
            0, 1, 2, 3, 4, 5, 6, 7, 8, 9, 10, 11, 12, 13, 14, 15);
        acc = __builtin_amdgcn_wmma_f32_16x16x32_f16(
            /*neg_a=*/false, afr[ky * 3 + kx], /*neg_b=*/false, bf,
            /*c_mod=*/(short)0, acc, /*reuse_a=*/false, /*reuse_b=*/false);
      }
    }
    // ---- epilogue: gate + fp32 store ----
    const int y   = y0 + row;
    const int ocb = g * OCG + mt * 16 + half * 8;
    #pragma unroll
    for (int r = 0; r < 8; ++r) {
      out[(((size_t)b * OUP + (ocb + r)) * H_ + y) * W_ + n] = acc[r] * gates[r];
    }
  }
}

// =====================================================================
// Launcher
// =====================================================================
extern "C" void kernel_launch(void* const* d_in, const int* in_sizes, int n_in,
                              void* d_out, int out_size, void* d_ws, size_t ws_size,
                              hipStream_t stream) {
  const float* x      = (const float*)d_in[0];
  const float* g1w1l  = (const float*)d_in[1];
  const float* g1w1g1 = (const float*)d_in[2];
  const float* g1w1g2 = (const float*)d_in[3];
  const float* g1w2l  = (const float*)d_in[4];
  const float* g1w2g  = (const float*)d_in[5];
  const float* g2w1l  = (const float*)d_in[6];
  const float* g2w1g1 = (const float*)d_in[7];
  const float* g2w1g2 = (const float*)d_in[8];
  const float* g2w2l  = (const float*)d_in[9];
  const float* g2w2g  = (const float*)d_in[10];
  float* out = (float*)d_out;

  // workspace layout
  _Float16* xsw = (_Float16*)d_ws;                       // 16*8*64*64*32 halfs = 33.5 MB
  float* wsf  = (float*)((char*)d_ws + (size_t)B_ * CGRP * H_ * W_ * ICG * sizeof(_Float16));
  float* psum = wsf;                                     // 8192*32 = 262144
  float* gap  = wsf + 262144;                            // 4096
  float* h1   = gap + 4096;                              // 4096
  float* g2b  = h1 + 4096;                               // 16
  float* cg   = g2b + 16;                                // 4096

  swizzle_gap_kernel<<<dim3(B_ * CGRP * H_), dim3(256), 0, stream>>>(x, xsw, psum);
  gap_reduce_kernel<<<dim3(B_), dim3(256), 0, stream>>>(psum, gap);
  glgf_kernel<<<dim3(B_), dim3(256), 0, stream>>>(
      gap, g1w1l, g1w1g1, g1w1g2, g1w2g,
      g2w1l, g2w1g1, g2w1g2, g2w2l, g2w2g,
      h1, g2b, cg);
  dynconv_wmma_kernel<<<dim3(B_ * CGRP * (H_ / RPB)), dim3(256), 0, stream>>>(
      xsw, g1w2l, h1, g2b, cg, out);
}
